// Otari_38027640439449
// MI455X (gfx1250) — matrix operand order repaired and checked by reading, not compiled, not run on hardware
//
#include <hip/hip_runtime.h>
#include <hip/hip_bf16.h>

// ---------------------------------------------------------------------------
// Types for CDNA5 WMMA (gfx1250, wave32)
// ---------------------------------------------------------------------------
typedef __attribute__((ext_vector_type(16))) __bf16 v16bf;
typedef __attribute__((ext_vector_type(8)))  __bf16 bf16x8;
typedef __attribute__((ext_vector_type(8)))  float  v8f;

__device__ __forceinline__ __bf16 f2bf(float f) {
  unsigned u = __builtin_bit_cast(unsigned, f);
  unsigned r = (u + 0x7FFFu + ((u >> 16) & 1u)) >> 16;  // round-to-nearest-even
  unsigned short s = (unsigned short)r;
  return __builtin_bit_cast(__bf16, s);
}

// Monotone encoding of float into uint for atomicMax-based segment max.
__device__ __forceinline__ unsigned f32_ord(float f) {
  unsigned u = __builtin_bit_cast(unsigned, f);
  return (u & 0x80000000u) ? ~u : (u | 0x80000000u);
}
__device__ __forceinline__ float ord_f32(unsigned u) {
  unsigned b = (u & 0x80000000u) ? (u & 0x7FFFFFFFu) : ~u;
  return __builtin_bit_cast(float, b);
}

// ---------------------------------------------------------------------------
// GEMM: C[M,N] = A[M,K] @ B[K,N] + bias[N] (+ res[M,N])
// A,B,C are f32 in memory; compute uses v_wmma_f32_16x16x32_bf16.
// Block: 256 threads (8 waves). Tile: BM=128, BN=128, BK=32.
// Wave grid 4x2 (M x N): each wave computes a 32x64 patch = 2x4 WMMA tiles.
// Requires N % 128 == 0 and K % 4 == 0 (true for every call here).
// ---------------------------------------------------------------------------
#define BM 128
#define BN 128
#define BK 32
#define LDS_STR 40   // 32 + 8 pad, in bf16 elements (row stride 80 B, 16B-aligned)

__global__ __launch_bounds__(256) void gemm_bf16_wmma(
    const float* __restrict__ A, const float* __restrict__ B,
    const float* __restrict__ bias, const float* __restrict__ res,
    float* __restrict__ Cout, int M, int N, int K)
{
  __shared__ alignas(16) __bf16 As[BM * LDS_STR];        // [row][k]
  __shared__ alignas(16) __bf16 Bs[BN * LDS_STR];        // [n][k] (transposed)

  const int tid  = threadIdx.x;
  const int lane = tid & 31;
  const int wid  = tid >> 5;
  const int wm   = wid >> 1;            // 0..3  : 32-row strip
  const int wn   = wid & 1;             // 0..1  : 64-col strip
  const int blockM = blockIdx.y * BM;
  const int blockN = blockIdx.x * BN;

  v8f acc[2][4];
  #pragma unroll
  for (int mi = 0; mi < 2; ++mi)
    #pragma unroll
    for (int ni = 0; ni < 4; ++ni)
      #pragma unroll
      for (int r = 0; r < 8; ++r) acc[mi][ni][r] = 0.0f;

  const int kb  = (lane & 16) >> 1;     // A-frag K sub-offset per ISA layout (0 or 8)
  const int bko = (lane & 16) >> 1;     // B-frag K offset (0 or 8)

  for (int k0 = 0; k0 < K; k0 += BK) {
    // ---- stage A tile: 128x32 f32 -> bf16 LDS ----
    #pragma unroll
    for (int i = 0; i < 4; ++i) {
      int f  = i * 256 + tid;           // 0..1023 float4 slots
      int r  = f >> 3;                  // row in tile
      int c  = (f & 7) << 2;            // k in tile (multiple of 4)
      int gr = blockM + r;
      int gk = k0 + c;
      float4 v = make_float4(0.f, 0.f, 0.f, 0.f);
      if (gr < M && gk < K) v = *(const float4*)(A + (size_t)gr * K + gk);
      __bf16* dstp = &As[r * LDS_STR + c];
      dstp[0] = f2bf(v.x); dstp[1] = f2bf(v.y);
      dstp[2] = f2bf(v.z); dstp[3] = f2bf(v.w);
    }
    // ---- stage B tile: 32x128 f32 -> bf16 LDS, transposed to [n][k] ----
    #pragma unroll
    for (int i = 0; i < 4; ++i) {
      int f  = i * 256 + tid;
      int r  = f >> 5;                  // k in tile
      int c  = (f & 31) << 2;           // n in tile (multiple of 4)
      int gk = k0 + r;
      float4 v = make_float4(0.f, 0.f, 0.f, 0.f);
      if (gk < K) v = *(const float4*)(B + (size_t)gk * N + (blockN + c));
      Bs[(c + 0) * LDS_STR + r] = f2bf(v.x);
      Bs[(c + 1) * LDS_STR + r] = f2bf(v.y);
      Bs[(c + 2) * LDS_STR + r] = f2bf(v.z);
      Bs[(c + 3) * LDS_STR + r] = f2bf(v.w);
    }
    // prefetch next K tile into cache hierarchy (global_prefetch_b8)
    if (k0 + BK < K) {
      int ra = tid >> 1;
      if (blockM + ra < M)
        __builtin_prefetch(A + (size_t)(blockM + ra) * K + (k0 + BK), 0, 0);
      if (k0 + BK + lane < K)
        __builtin_prefetch(B + (size_t)(k0 + BK + lane) * N + blockN, 0, 0);
    }
    __syncthreads();

    // ---- fragments + WMMA ----
    union Frag { v16bf v; bf16x8 h[2]; };
    Frag a[2], bf[4];
    const int arow0 = wm * 32 + (lane & 15);
    #pragma unroll
    for (int mi = 0; mi < 2; ++mi) {
      const __bf16* pA = &As[(arow0 + mi * 16) * LDS_STR];
      a[mi].h[0] = *reinterpret_cast<const bf16x8*>(pA + kb);
      a[mi].h[1] = *reinterpret_cast<const bf16x8*>(pA + 16 + kb);
    }
    const int bcol0 = wn * 64 + (lane & 15);
    #pragma unroll
    for (int ni = 0; ni < 4; ++ni) {
      const __bf16* pB = &Bs[(bcol0 + ni * 16) * LDS_STR + bko];
      bf[ni].h[0] = *reinterpret_cast<const bf16x8*>(pB);
      bf[ni].h[1] = *reinterpret_cast<const bf16x8*>(pB + 16);
    }
    #pragma unroll
    for (int mi = 0; mi < 2; ++mi)
      #pragma unroll
      for (int ni = 0; ni < 4; ++ni)
        acc[mi][ni] = __builtin_amdgcn_wmma_f32_16x16x32_bf16(
            false, a[mi].v, false, bf[ni].v, (short)0, acc[mi][ni], false, false);
    __syncthreads();
  }

  // ---- epilogue: bias (+ residual), store ----
  const int mhalf = (lane >> 4) << 3;   // C/D layout: lanes>=16 hold M+8
  const int ncol  = lane & 15;
  #pragma unroll
  for (int mi = 0; mi < 2; ++mi) {
    #pragma unroll
    for (int ni = 0; ni < 4; ++ni) {
      int nG = blockN + wn * 64 + ni * 16 + ncol;
      float bv = bias ? bias[nG] : 0.f;
      #pragma unroll
      for (int r = 0; r < 8; ++r) {
        int mG = blockM + wm * 32 + mi * 16 + mhalf + r;
        if (mG < M) {
          float v = acc[mi][ni][r] + bv;
          if (res) v += res[(size_t)mG * N + nG];
          Cout[(size_t)mG * N + nG] = v;
        }
      }
    }
  }
}

// ---------------------------------------------------------------------------
// Fill kernels
// ---------------------------------------------------------------------------
__global__ void fill_f32(float* p, float v, long long n) {
  long long i = (long long)blockIdx.x * blockDim.x + threadIdx.x;
  if (i < n) p[i] = v;
}
__global__ void fill_u32(unsigned* p, unsigned v, long long n) {
  long long i = (long long)blockIdx.x * blockDim.x + threadIdx.x;
  if (i < n) p[i] = v;
}

// ---------------------------------------------------------------------------
// GATv2 edge kernels (E real edges + Nn self loops)
// ---------------------------------------------------------------------------
__global__ void gat_edge_scores(const float* __restrict__ xl, const float* __restrict__ xr,
                                const int* __restrict__ src, const int* __restrict__ dst,
                                int E, int Nn, const float* __restrict__ att,
                                int H, int C, float* __restrict__ e_buf,
                                unsigned* __restrict__ m_enc)
{
  int edge = blockIdx.x * 8 + (threadIdx.x >> 5);
  int lane = threadIdx.x & 31;
  int Etot = E + Nn;
  if (edge >= Etot) return;
  int s, d;
  if (edge < E) { s = src[edge]; d = dst[edge]; } else { s = d = edge - E; }
  const int HC = H * C;
  for (int h = 0; h < H; ++h) {
    float acc = 0.f;
    const float* pl = xl + (size_t)s * HC + h * C;
    const float* pr = xr + (size_t)d * HC + h * C;
    const float* pa = att + h * C;
    for (int c = lane; c < C; c += 32) {
      float v = pl[c] + pr[c];
      v = (v > 0.f) ? v : 0.2f * v;              // leaky_relu(0.2)
      acc += v * pa[c];
    }
    #pragma unroll
    for (int off = 16; off; off >>= 1) acc += __shfl_xor(acc, off, 32);
    if (lane == 0) {
      e_buf[(size_t)edge * H + h] = acc;
      atomicMax(&m_enc[d * H + h], f32_ord(acc));
    }
  }
}

__global__ void gat_edge_exp(const float* __restrict__ e_buf, const int* __restrict__ dst,
                             int E, int Nn, int H, const unsigned* __restrict__ m_enc,
                             float* __restrict__ ex_buf, float* __restrict__ sum_buf)
{
  int idx = blockIdx.x * blockDim.x + threadIdx.x;   // over Etot*H
  int Etot = E + Nn;
  if (idx >= Etot * H) return;
  int edge = idx / H, h = idx - edge * H;
  int d = (edge < E) ? dst[edge] : (edge - E);
  float m = ord_f32(m_enc[d * H + h]);
  float ex = __expf(e_buf[idx] - m);
  ex_buf[idx] = ex;
  atomicAdd(&sum_buf[d * H + h], ex);
}

__global__ void gat_edge_agg(const float* __restrict__ xl, const float* __restrict__ ex_buf,
                             const int* __restrict__ src, const int* __restrict__ dst,
                             int E, int Nn, int H, int C, float* __restrict__ agg)
{
  long long idx = (long long)blockIdx.x * blockDim.x + threadIdx.x;
  const int HC = H * C;
  long long tot = (long long)(E + Nn) * HC;
  if (idx >= tot) return;
  int edge = (int)(idx / HC);
  int hc   = (int)(idx - (long long)edge * HC);
  int s, d;
  if (edge < E) { s = src[edge]; d = dst[edge]; } else { s = d = edge - E; }
  int h = hc / C;
  float w = ex_buf[(size_t)edge * H + h];
  atomicAdd(&agg[(size_t)d * HC + hc], w * xl[(size_t)s * HC + hc]);
}

// normalize by softmax sum, (optional) mean over 2 heads, + bias, BN(eval), ReLU
__global__ void gat_epilogue(const float* __restrict__ agg, const float* __restrict__ sum_buf,
                             const float* __restrict__ bias,
                             const float* __restrict__ bng, const float* __restrict__ bnb,
                             const float* __restrict__ bnm, const float* __restrict__ bnv,
                             int Nn, int H, int C, int mean_heads, float* __restrict__ out)
{
  int idx = blockIdx.x * blockDim.x + threadIdx.x;   // over Nn*C
  if (idx >= Nn * C) return;
  int i = idx / C, c = idx - i * C;
  float v;
  if (mean_heads) {  // H == 2, concat=False
    float v0 = agg[(size_t)i * H * C + c]     / sum_buf[i * H + 0];
    float v1 = agg[(size_t)i * H * C + C + c] / sum_buf[i * H + 1];
    v = 0.5f * (v0 + v1);
  } else {           // H == 1
    v = agg[(size_t)i * C + c] / sum_buf[i];
  }
  v += bias[c];
  v = (v - bnm[c]) * rsqrtf(bnv[c] + 1e-5f) * bng[c] + bnb[c];
  out[idx] = (v > 0.f) ? v : 0.f;
}

// ---------------------------------------------------------------------------
// LayerNorm: one wave per row
// ---------------------------------------------------------------------------
__global__ void layernorm_k(const float* __restrict__ in, const float* __restrict__ g,
                            const float* __restrict__ b, int Nn, int D,
                            float* __restrict__ out)
{
  int row  = blockIdx.x * 8 + (threadIdx.x >> 5);
  int lane = threadIdx.x & 31;
  if (row >= Nn) return;
  const float* p = in + (size_t)row * D;
  float s = 0.f, s2 = 0.f;
  for (int c = lane; c < D; c += 32) { float v = p[c]; s += v; s2 += v * v; }
  #pragma unroll
  for (int off = 16; off; off >>= 1) {
    s  += __shfl_xor(s,  off, 32);
    s2 += __shfl_xor(s2, off, 32);
  }
  float mu  = s / D;
  float var = s2 / D - mu * mu;
  float rs  = rsqrtf(var + 1e-5f);
  float* q = out + (size_t)row * D;
  for (int c = lane; c < D; c += 32) q[c] = (p[c] - mu) * rs * g[c] + b[c];
}

// ---------------------------------------------------------------------------
// Graph pooling (segment max) + MLP head
// ---------------------------------------------------------------------------
__global__ void pool_max(const float* __restrict__ feat, const int* __restrict__ batch,
                         int Nn, int D, unsigned* __restrict__ pool_enc)
{
  int idx = blockIdx.x * blockDim.x + threadIdx.x;   // over Nn*D
  if (idx >= Nn * D) return;
  int i = idx / D, c = idx - i * D;
  atomicMax(&pool_enc[batch[i] * D + c], f32_ord(feat[idx]));
}

__global__ void head_mlp(const unsigned* __restrict__ pool_enc,
                         const float* __restrict__ W1, const float* __restrict__ b1,
                         const float* __restrict__ W2, const float* __restrict__ b2,
                         float* __restrict__ out)
{
  __shared__ float pr[128], hh[128];
  int g = blockIdx.x, t = threadIdx.x;
  pr[t] = ord_f32(pool_enc[g * 128 + t]);
  __syncthreads();
  float a = b1[t];
  for (int k = 0; k < 128; ++k) a += pr[k] * W1[k * 128 + t];
  hh[t] = (a > 0.f) ? a : 0.f;
  __syncthreads();
  if (t < 30) {
    float o = b2[t];
    for (int j = 0; j < 128; ++j) o += hh[j] * W2[j * 30 + t];
    out[g * 30 + t] = o;
  }
}

// ---------------------------------------------------------------------------
// Host driver
// ---------------------------------------------------------------------------
extern "C" void kernel_launch(void* const* d_in, const int* in_sizes, int n_in,
                              void* d_out, int out_size, void* d_ws, size_t ws_size,
                              hipStream_t stream)
{
  (void)n_in; (void)out_size; (void)ws_size;
  const int Nn  = in_sizes[2];          // 10000 nodes
  const int E   = in_sizes[1] / 2;      // 160000 edges
  const int IN  = in_sizes[0] / Nn;     // 23600
  const int Etot = E + Nn;

  const float* x   = (const float*)d_in[0];
  const int* src   = (const int*)d_in[1];
  const int* dstp  = src + E;
  const int* batch = (const int*)d_in[2];
  auto P = [&](int i) { return (const float*)d_in[i]; };
  // params: d_in[3..62] = 6 GAT convs x {Wl,Wr,bl,br,att,b,bng,bnb,bnm,bnv}
  //         d_in[63..68] = ln1_g,ln1_b,ln2_g,ln2_b,ln3_g,ln3_b
  //         d_in[69..76] = l2_W,l2_b,l3_W,l3_b,h1_W,h1_b,h2_W,h2_b

  // workspace bump allocator (256B aligned)
  char* wsb = (char*)d_ws; size_t off = 0;
  auto alloc = [&](size_t bytes) -> void* {
    void* p = wsb + off; off = (off + bytes + 255) & ~(size_t)255; return p;
  };
  float*    xl       = (float*)alloc((size_t)Nn * 1024 * 4);
  float*    xr       = (float*)alloc((size_t)Nn * 1024 * 4);
  float*    agg      = (float*)alloc((size_t)Nn * 1024 * 4);
  float*    nodeA    = (float*)alloc((size_t)Nn * 512 * 4);
  float*    nodeB    = (float*)alloc((size_t)Nn * 512 * 4);
  float*    nodeC    = (float*)alloc((size_t)Nn * 512 * 4);
  float*    nodeD    = (float*)alloc((size_t)Nn * 512 * 4);
  float*    e_buf    = (float*)alloc((size_t)Etot * 2 * 4);
  float*    ex_buf   = (float*)alloc((size_t)Etot * 2 * 4);
  unsigned* m_enc    = (unsigned*)alloc((size_t)Nn * 2 * 4);
  float*    sum_buf  = (float*)alloc((size_t)Nn * 2 * 4);
  unsigned* pool_enc = (unsigned*)alloc(64 * 128 * 4);

  auto gemm = [&](const float* A, const float* W, const float* bias,
                  const float* res, float* C, int M, int N, int K) {
    dim3 g(N / 128, (M + 127) / 128);
    gemm_bf16_wmma<<<g, dim3(256), 0, stream>>>(A, W, bias, res, C, M, N, K);
  };
  auto fillF = [&](float* p, float v, long long n) {
    fill_f32<<<dim3((unsigned)((n + 255) / 256)), dim3(256), 0, stream>>>(p, v, n);
  };
  auto fillU = [&](unsigned* p, unsigned v, long long n) {
    fill_u32<<<dim3((unsigned)((n + 255) / 256)), dim3(256), 0, stream>>>(p, v, n);
  };
  auto run_gat = [&](const float* in, int K, int H, int C, int pb, float* outp) {
    int HC = H * C;
    gemm(in, P(pb + 0), P(pb + 2), nullptr, xl, Nn, HC, K);   // xl = in@Wl + bl
    gemm(in, P(pb + 1), P(pb + 3), nullptr, xr, Nn, HC, K);   // xr = in@Wr + br
    fillU(m_enc, 0u, (long long)Nn * H);                       // ord-encoded minimum
    fillF(sum_buf, 0.f, (long long)Nn * H);
    fillF(agg, 0.f, (long long)Nn * HC);
    gat_edge_scores<<<dim3((Etot + 7) / 8), dim3(256), 0, stream>>>(
        xl, xr, src, dstp, E, Nn, P(pb + 4), H, C, e_buf, m_enc);
    gat_edge_exp<<<dim3((Etot * H + 255) / 256), dim3(256), 0, stream>>>(
        e_buf, dstp, E, Nn, H, m_enc, ex_buf, sum_buf);
    long long aggTot = (long long)Etot * HC;
    gat_edge_agg<<<dim3((unsigned)((aggTot + 255) / 256)), dim3(256), 0, stream>>>(
        xl, ex_buf, src, dstp, E, Nn, H, C, agg);
    gat_epilogue<<<dim3((Nn * C + 255) / 256), dim3(256), 0, stream>>>(
        agg, sum_buf, P(pb + 5), P(pb + 6), P(pb + 7), P(pb + 8), P(pb + 9),
        Nn, H, C, (H == 2) ? 1 : 0, outp);
  };
  auto ln = [&](const float* in, int D, int gi, int bi, float* outp) {
    layernorm_k<<<dim3((Nn + 7) / 8), dim3(256), 0, stream>>>(in, P(gi), P(bi), Nn, D, outp);
  };

  // ---- Block 1 ----
  run_gat(x,     IN,  2, 512, 3,  nodeA);            // g11 -> [Nn,512]
  run_gat(nodeA, 512, 1, 512, 13, nodeB);            // g12 -> out1
  ln(nodeB, 512, 63, 64, nodeC);                     // lout1
  // ---- Block 2 ----
  run_gat(nodeC, 512, 2, 256, 23, nodeA);            // g21 -> [Nn,256]
  run_gat(nodeA, 256, 1, 256, 33, nodeB);            // g22 -> out2
  gemm(nodeC, P(69), P(70), nodeB, nodeD, Nn, 256, 512);  // rout2 = out2 + lout1@l2_W + b
  ln(nodeD, 256, 65, 66, nodeB);                     // lout2
  // ---- Block 3 ----
  run_gat(nodeB, 256, 2, 128, 43, nodeA);            // g31 -> [Nn,128]
  run_gat(nodeA, 128, 1, 128, 53, nodeC);            // g32 -> out3
  gemm(nodeB, P(71), P(72), nodeC, nodeD, Nn, 128, 256);  // rout3
  ln(nodeD, 128, 67, 68, nodeA);                     // lout3
  // ---- Pool + head ----
  fillU(pool_enc, 0u, 64 * 128);
  pool_max<<<dim3((Nn * 128 + 255) / 256), dim3(256), 0, stream>>>(
      nodeA, batch, Nn, 128, pool_enc);
  head_mlp<<<dim3(64), dim3(128), 0, stream>>>(
      pool_enc, P(73), P(74), P(75), P(76), (float*)d_out);
}
